// DeepSeekSparseAttention_44392781971820
// MI455X (gfx1250) — matrix operand order repaired and checked by reading, not compiled, not run on hardware
//
#include <hip/hip_runtime.h>
#include <hip/hip_bf16.h>
#include <stdint.h>

#define B_ 2
#define S_ 2048
#define D_ 1024
#define H_ 16
#define DK_ 64
#define HI_ 4
#define DI_ 64
#define TOPK_ 512

#define AS1 __attribute__((address_space(1)))
#define AS3 __attribute__((address_space(3)))

#if defined(__AMDGCN__) && __has_builtin(__builtin_amdgcn_global_load_async_to_lds_b128) && \
    __has_builtin(__builtin_amdgcn_s_wait_asynccnt)
#define HAS_ASYNC_LDS 1
#else
#define HAS_ASYNC_LDS 0
#endif

typedef __attribute__((ext_vector_type(16))) __bf16 v16bf;
typedef __attribute__((ext_vector_type(8)))  __bf16 v8bf;
typedef __attribute__((ext_vector_type(8)))  float  v8f;
typedef __attribute__((ext_vector_type(4)))  int    v4i;

static __device__ __forceinline__ v8f wmma_bf16f32(v16bf a, v16bf b, v8f c) {
  // D = A(16x32 bf16) * B(32x16 bf16) + C(16x16 f32)
  return __builtin_amdgcn_wmma_f32_16x16x32_bf16(false, a, false, b, (short)0, c,
                                                 false, false);
}
static __device__ __forceinline__ v16bf cat8(v8bf lo, v8bf hi) {
  return __builtin_shufflevector(lo, hi, 0,1,2,3,4,5,6,7,8,9,10,11,12,13,14,15);
}

// ---------------------------------------------------------------- casts
__global__ void cast_f32_bf16_kernel(const float* __restrict__ src,
                                     __bf16* __restrict__ dst, size_t n) {
  for (size_t i = (size_t)blockIdx.x * blockDim.x + threadIdx.x; i < n;
       i += (size_t)gridDim.x * blockDim.x)
    dst[i] = (__bf16)src[i];
}

// Wt[n*K + k] = W[k*N + n]   (f32 -> bf16)
__global__ void transpose_cast_kernel(const float* __restrict__ W,
                                      __bf16* __restrict__ Wt, int K, int N) {
  size_t total = (size_t)K * N;
  for (size_t i = (size_t)blockIdx.x * blockDim.x + threadIdx.x; i < total;
       i += (size_t)gridDim.x * blockDim.x) {
    int n = (int)(i / K), k = (int)(i % K);
    Wt[i] = (__bf16)W[(size_t)k * N + n];
  }
}

// ---------------------------------------------------------------- GEMM
// C(MxN,f32) = A(MxK,bf16 row-major) * Bt(NxK,bf16)^T
// block = 256 thr (8 waves), tile 128x128; wave = 64x32 (4x2 wmma tiles)
__global__ __launch_bounds__(256) void gemm_bf16_kernel(
    const __bf16* __restrict__ A, const __bf16* __restrict__ Bt,
    float* __restrict__ C, int M, int N, int K) {
  const int tid = threadIdx.x;
  const int wave = tid >> 5, lane = tid & 31;
  const int hi = lane >> 4, ln = lane & 15;
  const int wm0 = blockIdx.y * 128 + (wave >> 2) * 64;
  const int wn0 = blockIdx.x * 128 + (wave & 3) * 32;

  v8f acc[4][2];
#pragma unroll
  for (int i = 0; i < 4; i++)
#pragma unroll
    for (int j = 0; j < 2; j++) acc[i][j] = (v8f){0.f,0.f,0.f,0.f,0.f,0.f,0.f,0.f};

  for (int k0 = 0; k0 < K; k0 += 32) {
    v16bf afr[4], bfr[2];
#pragma unroll
    for (int i = 0; i < 4; i++) {
      int r = wm0 + i * 16 + ln; if (r >= M) r = M - 1;
      const __bf16* p = A + (size_t)r * K + k0 + hi * 16;
      __builtin_prefetch(p + 64, 0, 1);  // two k-steps ahead -> global_prefetch_b8
      afr[i] = *(const v16bf*)p;
    }
#pragma unroll
    for (int j = 0; j < 2; j++) {
      int n = wn0 + j * 16 + ln; if (n >= N) n = N - 1;
      const __bf16* p = Bt + (size_t)n * K + k0 + hi * 16;
      __builtin_prefetch(p + 64, 0, 1);
      bfr[j] = *(const v16bf*)p;
    }
#pragma unroll
    for (int i = 0; i < 4; i++)
#pragma unroll
      for (int j = 0; j < 2; j++) acc[i][j] = wmma_bf16f32(afr[i], bfr[j], acc[i][j]);
  }

  if ((wm0 + 63 < M) && (wn0 + 31 < N)) {
    // full tile: unguarded stores, 2 per row sharing one base address
#pragma unroll
    for (int i = 0; i < 4; i++)
#pragma unroll
      for (int e = 0; e < 8; e++) {
        float* cp = C + (size_t)(wm0 + i * 16 + hi * 8 + e) * N + wn0 + ln;
#pragma unroll
        for (int j = 0; j < 2; j++) cp[j * 16] = acc[i][j][e];
      }
  } else {
#pragma unroll
    for (int i = 0; i < 4; i++)
#pragma unroll
      for (int j = 0; j < 2; j++)
#pragma unroll
        for (int e = 0; e < 8; e++) {
          int r = wm0 + i * 16 + hi * 8 + e;
          int c = wn0 + j * 16 + ln;
          if (r < M && c < N) C[(size_t)r * N + c] = acc[i][j][e];
        }
  }
}

// ---------------------------------------------------------------- RoPE + pack
// qkv (B,S,3*D f32) -> Qb,Kb (B,H,S,64 bf16, roped), Vt (B,H,64,S bf16)
__global__ void rope_pack_kernel(const float* __restrict__ qkv,
                                 __bf16* __restrict__ Qb, __bf16* __restrict__ Kb,
                                 __bf16* __restrict__ Vt) {
  size_t idx = (size_t)blockIdx.x * blockDim.x + threadIdx.x;
  if (idx >= (size_t)B_ * S_ * H_ * 32) return;
  int i = idx & 31;
  int h = (int)((idx >> 5) & 15);
  int s = (int)((idx >> 9) & (S_ - 1));
  int b = (int)(idx >> 20);

  float theta = __powf(10000.f, -((float)(2 * i)) / 64.f);
  float sn, cs;
  __sincosf((float)s * theta, &sn, &cs);

  const float* row = qkv + (size_t)(b * S_ + s) * (3 * D_);
  size_t hs = ((size_t)(b * H_ + h) * S_ + s) * DK_;

  float x1 = row[h * 64 + 2 * i], x2 = row[h * 64 + 2 * i + 1];
  Qb[hs + 2 * i]     = (__bf16)(x1 * cs - x2 * sn);
  Qb[hs + 2 * i + 1] = (__bf16)(x2 * cs + x1 * sn);

  x1 = row[D_ + h * 64 + 2 * i]; x2 = row[D_ + h * 64 + 2 * i + 1];
  Kb[hs + 2 * i]     = (__bf16)(x1 * cs - x2 * sn);
  Kb[hs + 2 * i + 1] = (__bf16)(x2 * cs + x1 * sn);

  float v1 = row[2 * D_ + h * 64 + 2 * i], v2 = row[2 * D_ + h * 64 + 2 * i + 1];
  __bf16* vb = Vt + (size_t)(b * H_ + h) * DK_ * S_;
  vb[(size_t)(2 * i) * S_ + s]     = (__bf16)v1;
  vb[(size_t)(2 * i + 1) * S_ + s] = (__bf16)v2;
}

// ---------------------------------------------------------------- indexer scores
// scores[b][t][s] = sum_h relu(qi[b,t,h,:]·ki[b,s,:]) * w[b,t,h]   (causal -1e9)
// 1 wave per 16x16 (t,s) tile
__global__ __launch_bounds__(32) void index_score_kernel(
    const __bf16* __restrict__ qib, const __bf16* __restrict__ kib,
    const float* __restrict__ w, float* __restrict__ scores) {
  const int sblk = blockIdx.x * 16, tblk = blockIdx.y * 16, b = blockIdx.z;
  const int lane = threadIdx.x;
  const int hi = lane >> 4, ln = lane & 15;

  if (sblk > tblk + 15) {  // entirely above diagonal
#pragma unroll
    for (int i = 0; i < 8; i++) {
      int t = tblk + hi * 8 + i;
      scores[((size_t)(b * S_ + t)) * S_ + sblk + ln] = -1.0e9f;
    }
    return;
  }

  __shared__ float wl[64];
  wl[lane] = w[(size_t)(b * S_ + tblk) * HI_ + lane];
  wl[lane + 32] = w[(size_t)(b * S_ + tblk) * HI_ + lane + 32];
  __syncthreads();

  v8f acc = (v8f){0.f,0.f,0.f,0.f,0.f,0.f,0.f,0.f};
#pragma unroll
  for (int h = 0; h < HI_; h++) {
    const __bf16* qp = qib + ((size_t)(b * S_ + tblk + ln) * HI_ + h) * DI_;
    v16bf a0 = *(const v16bf*)(qp + hi * 16);
    v16bf a1 = *(const v16bf*)(qp + 32 + hi * 16);
    const __bf16* kp = kib + (size_t)(b * S_ + sblk + ln) * DI_ + hi * 16;
    v16bf b0 = *(const v16bf*)kp;
    v16bf b1 = *(const v16bf*)(kp + 32);
    v8f d = (v8f){0.f,0.f,0.f,0.f,0.f,0.f,0.f,0.f};
    d = wmma_bf16f32(a0, b0, d);
    d = wmma_bf16f32(a1, b1, d);
#pragma unroll
    for (int i = 0; i < 8; i++)
      acc[i] += fmaxf(d[i], 0.f) * wl[(hi * 8 + i) * HI_ + h];
  }
#pragma unroll
  for (int i = 0; i < 8; i++) {
    int t = tblk + hi * 8 + i;
    int s = sblk + ln;
    scores[((size_t)(b * S_ + t)) * S_ + s] = (s <= t) ? acc[i] : -1.0e9f;
  }
}

// ---------------------------------------------------------------- exact top-512
// one block per (b,t) row; radix-select exact threshold, ties by lowest index.
__global__ __launch_bounds__(256) void topk_mask_kernel(
    const float* __restrict__ scores, unsigned* __restrict__ mask) {
  const int row = blockIdx.x;  // b*S + t
  const int tid = threadIdx.x;
  const float* sp = scores + (size_t)row * S_;

  __shared__ unsigned keys[S_];
  __shared__ unsigned hist[256];
  __shared__ unsigned scn[256];
  __shared__ unsigned wbuf[S_ / 32];
  __shared__ unsigned sh_prefix;
  __shared__ int sh_k;

#pragma unroll
  for (int j = 0; j < 8; j++) {
    int s = tid * 8 + j;
    unsigned u = __float_as_uint(sp[s]);
    u = (u & 0x80000000u) ? ~u : (u | 0x80000000u);  // order-preserving flip
    keys[s] = u;
  }
  __syncthreads();

  unsigned prefix = 0;
  int k = TOPK_;
  for (int pass = 3; pass >= 0; pass--) {
    hist[tid] = 0;
    __syncthreads();
    int shift = pass * 8;
#pragma unroll
    for (int j = 0; j < 8; j++) {
      unsigned key = keys[tid * 8 + j];
      unsigned high = (pass == 3) ? 0u : (key >> (shift + 8));
      if (high == prefix) atomicAdd(&hist[(key >> shift) & 255u], 1u);
    }
    __syncthreads();
    if (tid == 0) {
      int cum = 0, kk = k;
      unsigned chosen = 0;
      for (int bb = 255; bb >= 0; bb--) {
        int c = (int)hist[bb];
        if (cum + c >= kk) { chosen = (unsigned)bb; kk -= cum; break; }
        cum += c;
      }
      sh_prefix = (prefix << 8) | chosen;
      sh_k = kk;
    }
    __syncthreads();
    prefix = sh_prefix;
    k = sh_k;
    __syncthreads();
  }
  const unsigned T = prefix;   // exact 512th-largest key
  const int mrem = k;          // # ties (==T) to take, lowest index first

  int tcnt = 0;
#pragma unroll
  for (int j = 0; j < 8; j++) tcnt += (keys[tid * 8 + j] == T) ? 1 : 0;
  scn[tid] = (unsigned)tcnt;
  __syncthreads();
  for (int off = 1; off < 256; off <<= 1) {
    unsigned v = (tid >= off) ? scn[tid - off] : 0u;
    __syncthreads();
    scn[tid] += v;
    __syncthreads();
  }
  int rank = (int)scn[tid] - tcnt;  // exclusive tie rank at this thread's start
  __syncthreads();

  unsigned byte = 0;
#pragma unroll
  for (int j = 0; j < 8; j++) {
    unsigned key = keys[tid * 8 + j];
    bool sel = (key > T) || (key == T && rank < mrem);
    if (key == T) rank++;
    if (sel) byte |= (1u << j);
  }
  if (tid < S_ / 32) wbuf[tid] = 0;
  __syncthreads();
  atomicOr(&wbuf[tid >> 2], byte << ((tid & 3) * 8));
  __syncthreads();
  if (tid < S_ / 32) mask[(size_t)row * (S_ / 32) + tid] = wbuf[tid];
}

// ---------------------------------------------------------------- masked flash attention
// grid (S/128, H, B), 256 thr (8 waves); wave owns 16 t-rows; dense s-loop with
// top-k bitmask from LDS (async-copied); WMMA for QK^T and P*V.
__global__ __launch_bounds__(256) void flash_kernel(
    const __bf16* __restrict__ Qb, const __bf16* __restrict__ Kb,
    const __bf16* __restrict__ Vt, const unsigned* __restrict__ mask,
    float* __restrict__ attn) {
  const int tblk = blockIdx.x, h = blockIdx.y, b = blockIdx.z;
  const int tid = threadIdx.x;
  const int wave = tid >> 5, lane = tid & 31;
  const int hi = lane >> 4, ln = lane & 15;
  const int t0blk = tblk * 128;

  __shared__ unsigned mlds[128 * (S_ / 32)];            // 32 KB
  __shared__ __align__(32) __bf16 plds[8 * 16 * 32];    // 8 KB (1KB per wave)

  {
    const unsigned* gsrc = mask + (size_t)(b * S_ + t0blk) * (S_ / 32);
#if HAS_ASYNC_LDS
    // 32KB block copy via ASYNCcnt path, no VGPR round-trip
#pragma unroll
    for (int c = 0; c < 8; c++) {
      unsigned boff = (unsigned)(tid + c * 256) * 16u;
      AS1 v4i* gp = (AS1 v4i*)((size_t)gsrc + boff);
      AS3 v4i* lp = (AS3 v4i*)((unsigned)(size_t)(const void*)mlds + boff);
      __builtin_amdgcn_global_load_async_to_lds_b128(gp, lp, 0, 0);
    }
    __builtin_amdgcn_s_wait_asynccnt(0);
#else
    for (int i = tid; i < 128 * (S_ / 32); i += 256) mlds[i] = gsrc[i];
#endif
  }
  __syncthreads();

  const int t0 = t0blk + wave * 16;
  const __bf16* Qrow = Qb + ((size_t)(b * H_ + h) * S_ + t0) * DK_;
  const __bf16* Kbase = Kb + (size_t)(b * H_ + h) * S_ * DK_;
  const __bf16* Vbase = Vt + (size_t)(b * H_ + h) * DK_ * S_;

  const __bf16* qp = Qrow + (size_t)ln * DK_;
  v16bf qa0 = *(const v16bf*)(qp + hi * 16);
  v16bf qa1 = *(const v16bf*)(qp + 32 + hi * 16);

  v8f o[4];
#pragma unroll
  for (int dn = 0; dn < 4; dn++) o[dn] = (v8f){0.f,0.f,0.f,0.f,0.f,0.f,0.f,0.f};
  float m[8], lsum[8];
#pragma unroll
  for (int i = 0; i < 8; i++) { m[i] = -1e30f; lsum[i] = 0.f; }

  __bf16* pw = plds + wave * (16 * 32);
  const float NEG = -1e30f;
  int send = t0blk + 127; if (send < TOPK_ - 1) send = TOPK_ - 1;

  for (int s0 = 0; s0 <= send; s0 += 32) {
    // ---- logits (two 16-col halves), K = 64 via two wmma steps
    float pbuf[2][8];
    float tmax[8];
#pragma unroll
    for (int i = 0; i < 8; i++) tmax[i] = NEG;
#pragma unroll
    for (int half = 0; half < 2; half++) {
      int sh = s0 + half * 16;
      const __bf16* kp = Kbase + (size_t)(sh + ln) * DK_ + hi * 16;
      __builtin_prefetch(kp + 32 * DK_, 0, 1);  // next s-tile
      v16bf b0 = *(const v16bf*)kp;
      v16bf b1 = *(const v16bf*)(kp + 32);
      v8f c = (v8f){0.f,0.f,0.f,0.f,0.f,0.f,0.f,0.f};
      c = wmma_bf16f32(qa0, b0, c);
      c = wmma_bf16f32(qa1, b1, c);
      int col = sh + ln;
      int wi = col >> 5, bit = col & 31;
#pragma unroll
      for (int i = 0; i < 8; i++) {
        int rl = wave * 16 + hi * 8 + i;
        unsigned wm = mlds[rl * (S_ / 32) + wi];
        float v = ((wm >> bit) & 1u) ? c[i] * 0.125f : NEG;
        pbuf[half][i] = v;
        tmax[i] = fmaxf(tmax[i], v);
      }
    }
    // ---- row max across the 16 lanes of each half-wave
#pragma unroll
    for (int off = 1; off < 16; off <<= 1)
#pragma unroll
      for (int i = 0; i < 8; i++)
        tmax[i] = fmaxf(tmax[i], __shfl_xor(tmax[i], off, 16));

    float resc[8], rsum[8];
#pragma unroll
    for (int i = 0; i < 8; i++) {
      float nm = fmaxf(m[i], tmax[i]);
      resc[i] = __expf(m[i] - nm);
      m[i] = nm;
      rsum[i] = 0.f;
    }
#pragma unroll
    for (int half = 0; half < 2; half++)
#pragma unroll
      for (int i = 0; i < 8; i++) {
        float p = __expf(pbuf[half][i] - m[i]);
        pbuf[half][i] = p;
        rsum[i] += p;
      }
#pragma unroll
    for (int off = 1; off < 16; off <<= 1)
#pragma unroll
      for (int i = 0; i < 8; i++) rsum[i] += __shfl_xor(rsum[i], off, 16);
#pragma unroll
    for (int i = 0; i < 8; i++) lsum[i] = lsum[i] * resc[i] + rsum[i];
#pragma unroll
    for (int dn = 0; dn < 4; dn++)
#pragma unroll
      for (int i = 0; i < 8; i++) o[dn][i] *= resc[i];

    // ---- P (C-layout) -> LDS -> A-layout fragment
#pragma unroll
    for (int half = 0; half < 2; half++)
#pragma unroll
      for (int i = 0; i < 8; i++)
        pw[(hi * 8 + i) * 32 + half * 16 + ln] = (__bf16)pbuf[half][i];
    const __bf16* pr = pw + ln * 32 + hi * 16;
    v16bf pa = *(const v16bf*)pr;

    // ---- O += P(16x32) * V(32x64) : 4 wmma over d tiles
#pragma unroll
    for (int dn = 0; dn < 4; dn++) {
      const __bf16* vp = Vbase + (size_t)(dn * 16 + ln) * S_ + s0 + hi * 16;
      __builtin_prefetch(vp + 32, 0, 1);  // next s-tile
      v16bf vb = *(const v16bf*)vp;
      o[dn] = wmma_bf16f32(pa, vb, o[dn]);
    }
  }

#pragma unroll
  for (int i = 0; i < 8; i++) {
    float inv = 1.0f / lsum[i];
#pragma unroll
    for (int dn = 0; dn < 4; dn++) o[dn][i] *= inv;
  }
#pragma unroll
  for (int dn = 0; dn < 4; dn++)
#pragma unroll
    for (int i = 0; i < 8; i++) {
      int t = t0 + hi * 8 + i;
      attn[(size_t)(b * S_ + t) * D_ + h * DK_ + dn * 16 + ln] = o[dn][i];
    }
}

// ---------------------------------------------------------------- launch
extern "C" void kernel_launch(void* const* d_in, const int* in_sizes, int n_in,
                              void* d_out, int out_size, void* d_ws, size_t ws_size,
                              hipStream_t stream) {
  const float* x      = (const float*)d_in[0];
  const float* Wqkv   = (const float*)d_in[1];
  const float* Wo     = (const float*)d_in[2];
  const float* Wq_idx = (const float*)d_in[3];
  const float* Wk_idx = (const float*)d_in[4];
  const float* Ww_idx = (const float*)d_in[5];
  float* out = (float*)d_out;

  char* ws = (char*)d_ws;
  size_t off = 0;
  auto alloc = [&](size_t bytes) -> void* {
    void* p = ws + off;
    off += (bytes + 255) & ~(size_t)255;
    return p;
  };
  const size_t MS = (size_t)B_ * S_;  // 4096 rows

  __bf16* xb     = (__bf16*)alloc(MS * D_ * 2);
  __bf16* Wqkvt  = (__bf16*)alloc((size_t)3 * D_ * D_ * 2);
  __bf16* Wqit   = (__bf16*)alloc((size_t)HI_ * DI_ * D_ * 2);
  __bf16* Wkit   = (__bf16*)alloc((size_t)DI_ * D_ * 2);
  __bf16* Wwit   = (__bf16*)alloc((size_t)HI_ * D_ * 2);
  __bf16* Wot    = (__bf16*)alloc((size_t)D_ * D_ * 2);
  float*  qkv    = (float*) alloc(MS * 3 * D_ * 4);
  float*  qi     = (float*) alloc(MS * HI_ * DI_ * 4);
  float*  ki     = (float*) alloc(MS * DI_ * 4);
  float*  wbuf   = (float*) alloc(MS * HI_ * 4);
  __bf16* qib    = (__bf16*)alloc(MS * HI_ * DI_ * 2);
  __bf16* kib    = (__bf16*)alloc(MS * DI_ * 2);
  __bf16* Qb     = (__bf16*)alloc((size_t)B_ * H_ * S_ * DK_ * 2);
  __bf16* Kb     = (__bf16*)alloc((size_t)B_ * H_ * S_ * DK_ * 2);
  __bf16* Vt     = (__bf16*)alloc((size_t)B_ * H_ * DK_ * S_ * 2);
  float*  scores = (float*) alloc(MS * S_ * 4);
  unsigned* msk  = (unsigned*)alloc(MS * (S_ / 32) * 4);
  float*  attn   = (float*) alloc(MS * D_ * 4);
  __bf16* attnb  = (__bf16*)alloc(MS * D_ * 2);

  // 1. pack operands to bf16 (A row-major, weights transposed to N x K)
  cast_f32_bf16_kernel<<<1024, 256, 0, stream>>>(x, xb, MS * D_);
  transpose_cast_kernel<<<2048, 256, 0, stream>>>(Wqkv,   Wqkvt, D_, 3 * D_);
  transpose_cast_kernel<<<512,  256, 0, stream>>>(Wq_idx, Wqit,  D_, HI_ * DI_);
  transpose_cast_kernel<<<256,  256, 0, stream>>>(Wk_idx, Wkit,  D_, DI_);
  transpose_cast_kernel<<<32,   256, 0, stream>>>(Ww_idx, Wwit,  D_, HI_);
  transpose_cast_kernel<<<2048, 256, 0, stream>>>(Wo,     Wot,   D_, D_);

  // 2. projection GEMMs
  gemm_bf16_kernel<<<dim3(24, 32), 256, 0, stream>>>(xb, Wqkvt, qkv,
                                                     (int)MS, 3 * D_, D_);
  gemm_bf16_kernel<<<dim3(2, 32), 256, 0, stream>>>(xb, Wqit, qi,
                                                    (int)MS, HI_ * DI_, D_);
  gemm_bf16_kernel<<<dim3(1, 32), 256, 0, stream>>>(xb, Wkit, ki,
                                                    (int)MS, DI_, D_);
  gemm_bf16_kernel<<<dim3(1, 32), 256, 0, stream>>>(xb, Wwit, wbuf,
                                                    (int)MS, HI_, D_);
  cast_f32_bf16_kernel<<<1024, 256, 0, stream>>>(qi, qib, MS * HI_ * DI_);
  cast_f32_bf16_kernel<<<512, 256, 0, stream>>>(ki, kib, MS * DI_);

  // 3. RoPE + layout packs for attention
  rope_pack_kernel<<<(unsigned)((MS * H_ * 32 + 255) / 256), 256, 0, stream>>>(
      qkv, Qb, Kb, Vt);

  // 4. lightning indexer scores (WMMA) with causal -1e9 mask
  index_score_kernel<<<dim3(S_ / 16, S_ / 16, B_), 32, 0, stream>>>(
      qib, kib, wbuf, scores);

  // 5. exact top-512 per row -> selection bitmask
  topk_mask_kernel<<<(unsigned)MS, 256, 0, stream>>>(scores, msk);

  // 6. masked flash attention (WMMA)
  flash_kernel<<<dim3(S_ / 128, H_, B_), 256, 0, stream>>>(Qb, Kb, Vt, msk, attn);

  // 7. output projection
  cast_f32_bf16_kernel<<<1024, 256, 0, stream>>>(attn, attnb, MS * D_);
  gemm_bf16_kernel<<<dim3(8, 32), 256, 0, stream>>>(attnb, Wot, out,
                                                    (int)MS, D_, D_);
}
